// SparseAttention_8933531975741
// MI455X (gfx1250) — matrix operand order repaired and checked by reading, not compiled, not run on hardware
//
#include <hip/hip_runtime.h>
#include <hip/hip_bf16.h>

// CDNA5 / gfx1250 sparse (top-k causal) attention.
// Shapes fixed by the reference: B=8, L=S=1024, H=8, E=D=64, k=307.
// 4 waves (128 threads) per 16-query tile cooperating through a shared
// 64 KB LDS score buffer; WMMA f16->f32 for both matmuls; top-k selection
// done on register-resident sign-remapped keys (single LDS read per row).

typedef __attribute__((ext_vector_type(16))) _Float16 v16h;
typedef __attribute__((ext_vector_type(8)))  float    v8f;

#define B_    8
#define L_    1024
#define S_    1024
#define H_    8
#define E_    64
#define D_    64
#define TQ    16
#define NW    4     // waves per workgroup
#define KTOP  307   // max(1, int(1024 * 0.3))
#define CAP   (S_ / 32)   // row elements per lane (= 32)

__device__ __forceinline__ float wave_max(float v) {
#pragma unroll
    for (int o = 16; o > 0; o >>= 1) v = fmaxf(v, __shfl_xor(v, o, 32));
    return v;
}
__device__ __forceinline__ unsigned wave_max_u(unsigned v) {
#pragma unroll
    for (int o = 16; o > 0; o >>= 1) {
        unsigned w = (unsigned)__shfl_xor((int)v, o, 32);
        v = (w > v) ? w : v;
    }
    return v;
}
__device__ __forceinline__ float wave_sum(float v) {
#pragma unroll
    for (int o = 16; o > 0; o >>= 1) v += __shfl_xor(v, o, 32);
    return v;
}
__device__ __forceinline__ int wave_sum_i(int v) {
#pragma unroll
    for (int o = 16; o > 0; o >>= 1) v += __shfl_xor(v, o, 32);
    return v;
}

// Map float bits to monotone-increasing unsigned order (and back).
__device__ __forceinline__ unsigned f2ord(unsigned x) {
    return x ^ ((x & 0x80000000u) ? 0xFFFFFFFFu : 0x80000000u);
}
__device__ __forceinline__ unsigned ord2f(unsigned u) {
    return (u & 0x80000000u) ? (u ^ 0x80000000u) : ~u;
}

// A-operand (f16 16x32) K-offset for VGPR-half v (0..7), lane group g (0..1).
__device__ __forceinline__ int a_koff(int v, int g) {
    return (v < 4) ? (2 * v + 8 * g) : (16 + 2 * (v - 4) + 8 * g);
}

__global__ __launch_bounds__(NW * 32)
void sparse_attn_gfx1250(const float* __restrict__ Q,
                         const float* __restrict__ K,
                         const float* __restrict__ V,
                         float* __restrict__ O) {
    // 16 query rows x full S of fp32 scores -> exactly 64 KB of LDS,
    // plus a 4 KB fp32 output staging tile for cross-wave reduction.
    __shared__ float sP[TQ * S_];
    __shared__ float sO[TQ * D_];

    const int tid  = (int)threadIdx.x;   // 0..127
    const int lane = tid & 31;           // lane within wave32
    const int wid  = tid >> 5;           // wave id 0..3
    const int blk  = (int)blockIdx.x;    // query tile index (0..63)
    const int h    = (int)blockIdx.y;
    const int b    = (int)blockIdx.z;

    const int q0 = blk * TQ;
    const int n  = lane & 15;            // N index (C/D layout) / M index (A layout)
    const int g  = lane >> 4;            // lane group
    const int M  = n;                    // A-layout row held by this lane
    const int krow = n + 16 * g;         // V row in a 32-key chunk this lane prefetches (0..31)

    const int ncols     = (blk + 1) * TQ;          // causally reachable keys
    const int ncols_pad = (ncols + 31) & ~31;      // padded to WMMA K=32 chunks
    const float scale   = 0.125f;                  // 1/sqrt(64)

    // ---------------- Load Q tile into WMMA A-layout (f16), E=64 -> 2 chunks.
    v16h qa[2];
    {
        const float* qrow = Q + (((size_t)b * L_ + (size_t)(q0 + M)) * H_ + h) * E_;
#pragma unroll
        for (int ec = 0; ec < 2; ++ec) {
#pragma unroll
            for (int v = 0; v < 8; ++v) {
                int kb = a_koff(v, g);
                qa[ec][2 * v]     = (_Float16)qrow[ec * 32 + kb];
                qa[ec][2 * v + 1] = (_Float16)qrow[ec * 32 + kb + 1];
            }
        }
    }

    // ---------------- Phase 1: masked scores S = Q K^T into LDS.
    // Key tiles round-robined across the 4 waves (disjoint LDS columns).
    for (int j = wid; j <= blk; j += NW) {
        v8f c = {};
#pragma unroll
        for (int ec = 0; ec < 2; ++ec) {
            // B operand: 32x16 f16; lane holds key j*16+n, halves = 16 E values (a_koff order).
            // For fixed nn the 32 lanes read 16 key rows (2 floats each), H*E floats apart.
            v16h kb;
#pragma unroll
            for (int nn = 0; nn < 16; ++nn) {
                kb[nn] = (_Float16)K[(((size_t)b * S_ + (size_t)(j * 16 + n)) * H_ + h) * E_
                                     + ec * 32 + a_koff(nn >> 1, g) + (nn & 1)];
            }
            c = __builtin_amdgcn_wmma_f32_16x16x32_f16(
                    false, qa[ec], false, kb, (short)0, c, false, false);
        }
        // C/D layout: lane holds D[v + 8*g][n]; apply causal mask, spill to LDS.
#pragma unroll
        for (int v = 0; v < 8; ++v) {
            int row = v + 8 * g;
            int key = j * 16 + n;
            float s = c[v];
            if (key > q0 + row) s = -__builtin_inff();
            sP[row * S_ + key] = s;
        }
    }
    __syncthreads();

    // ---------------- Phase 2/3: per-row top-k threshold + normalized softmax.
    // Each wave owns 4 rows. The row is read from LDS exactly once into a
    // register-resident array of sortable keys; all 32 radix passes are VALU.
    for (int r = wid * (TQ / NW); r < (wid + 1) * (TQ / NW); ++r) {
        const int q      = q0 + r;
        const int nvalid = q + 1;                  // causal row length
        float* row = &sP[r * S_];

        // Single LDS sweep: remapped keys; 0 sentinel == "below everything".
        unsigned u[CAP];
#pragma unroll
        for (int i = 0; i < CAP; ++i) {
            const int cidx = lane + i * 32;
            unsigned uu = 0u;
            if (cidx < nvalid) uu = f2ord(__float_as_uint(row[cidx]));
            u[i] = uu;
        }

        // Row max (top-k always contains the max).
        unsigned umax = 0u;
#pragma unroll
        for (int i = 0; i < CAP; ++i) umax = (u[i] > umax) ? u[i] : umax;
        umax = wave_max_u(umax);
        const float m = __uint_as_float(ord2f(umax));

        // Exact k-th largest via 32-step radix binary search (pure VALU:
        // sentinel 0 never counts since every candidate has a bit set).
        unsigned res = 0u;
        if (nvalid > KTOP) {
            for (int bit = 31; bit >= 0; --bit) {
                const unsigned cand = res | (1u << bit);
                int cnt = 0;
#pragma unroll
                for (int i = 0; i < CAP; ++i) cnt += (u[i] >= cand) ? 1 : 0;
                cnt = wave_sum_i(cnt);
                if (cnt >= KTOP) res = cand;
            }
        }

        // One exp per kept element into registers + denominator.
        float pr[CAP];
        float ssum = 0.0f;
#pragma unroll
        for (int i = 0; i < CAP; ++i) {
            const int cidx = lane + i * 32;
            float p = 0.0f;
            if (cidx < nvalid && u[i] >= res) {
                const float s = __uint_as_float(ord2f(u[i]));
                p = __expf(scale * (s - m));
            }
            pr[i] = p;
            ssum += p;
        }
        ssum = wave_sum(ssum);
        const float inv = 1.0f / ssum;

        // Single LDS write sweep: pre-normalized probabilities (padded cols 0).
#pragma unroll
        for (int i = 0; i < CAP; ++i) {
            const int cidx = lane + i * 32;
            if (cidx < ncols_pad) row[cidx] = pr[i] * inv;
        }
    }

    // Zero the output staging tile (16x64 fp32).
#pragma unroll
    for (int i = 0; i < (TQ * D_) / (NW * 32); ++i)
        sO[tid + i * (NW * 32)] = 0.0f;
    __syncthreads();

    // ---------------- Phase 4: O = P * V via WMMA, D=64 -> 4 N-tiles.
    // Key chunks round-robined across waves; private accumulators reduced
    // into sO with LDS float atomics (ds_add_f32).
    v8f acc[4] = {v8f{}, v8f{}, v8f{}, v8f{}};
    const int nchunks = ncols_pad / 32;
    for (int t = wid; t < nchunks; t += NW) {
        const int kb0 = t * 32;

        // Prefetch this lane's V row for the next chunk this wave will touch.
        if (t + NW < nchunks) {
            const float* vnext =
                V + (((size_t)b * S_ + (size_t)((t + NW) * 32 + krow)) * H_ + h) * D_;
            __builtin_prefetch(vnext, 0, 0);
        }

        // A operand: probabilities in A-layout (lane row M, f16).
        v16h pa;
#pragma unroll
        for (int v = 0; v < 8; ++v) {
            int kk = a_koff(v, g);
            pa[2 * v]     = (_Float16)sP[M * S_ + kb0 + kk];
            pa[2 * v + 1] = (_Float16)sP[M * S_ + kb0 + kk + 1];
        }

        // B operand: V; lane holds D column n, halves = keys kb0 + a_koff order.
        const float* vcol =
            V + (((size_t)b * S_ + (size_t)kb0) * H_ + h) * D_ + n;
#pragma unroll
        for (int nt = 0; nt < 4; ++nt) {
            v16h vb;
#pragma unroll
            for (int s = 0; s < 16; ++s) {
                int kk = a_koff(s >> 1, g) + (s & 1);
                vb[s] = (_Float16)vcol[(size_t)kk * H_ * D_ + nt * 16];
            }
            acc[nt] = __builtin_amdgcn_wmma_f32_16x16x32_f16(
                          false, pa, false, vb, (short)0, acc[nt], false, false);
        }
    }

    // Cross-wave reduction of the 16x64 accumulator tile.
#pragma unroll
    for (int nt = 0; nt < 4; ++nt) {
#pragma unroll
        for (int v = 0; v < 8; ++v) {
            int row = v + 8 * g;
            atomicAdd(&sO[row * D_ + nt * 16 + n], acc[nt][v]);
        }
    }
    __syncthreads();

    // ---------------- Coalesced writeback: thread t stores 8 contiguous floats.
    {
        const int row  = tid >> 3;          // 0..15
        const int col8 = (tid & 7) * 8;     // 0,8,...,56
        const float4* src = (const float4*)&sO[row * D_ + col8];
        float4* dst = (float4*)
            (O + (((size_t)b * L_ + (size_t)(q0 + row)) * H_ + h) * D_ + col8);
        dst[0] = src[0];
        dst[1] = src[1];
    }
}

extern "C" void kernel_launch(void* const* d_in, const int* in_sizes, int n_in,
                              void* d_out, int out_size, void* d_ws, size_t ws_size,
                              hipStream_t stream) {
    (void)in_sizes; (void)n_in; (void)d_ws; (void)ws_size; (void)out_size;
    const float* Q = (const float*)d_in[0];   // [B,L,H,E] fp32
    const float* K = (const float*)d_in[1];   // [B,S,H,E] fp32
    const float* V = (const float*)d_in[2];   // [B,S,H,D] fp32
    // d_in[3] is the boolean causal mask; the kernel applies causality analytically.
    float* O = (float*)d_out;                 // [B,L,H,D] fp32

    dim3 grid(L_ / TQ, H_, B_);               // 64 x 8 x 8 = 4096 workgroups
    dim3 block(NW * 32, 1, 1);                // 4 waves per 16-query tile
    sparse_attn_gfx1250<<<grid, block, 0, stream>>>(Q, K, V, O);
}
